// MoE_20444044329517
// MI455X (gfx1250) — matrix operand order repaired and checked by reading, not compiled, not run on hardware
//
#include <hip/hip_runtime.h>
#include <hip/hip_bf16.h>
#include <cstdint>

#define D_MODEL 1024
#define NEXP    8
#define HID     2752     // 43 * 64, 86 * 32
#define NTOK    8192
#define ALPHA_LB 0.05f

typedef __attribute__((ext_vector_type(16))) __bf16 v16bf;
typedef __attribute__((ext_vector_type(8)))  float  v8f;
typedef __attribute__((ext_vector_type(4)))  int    v4i;
typedef __attribute__((address_space(1))) v4i* gv4i_p;   // global int4*
typedef __attribute__((address_space(3))) v4i* lv4i_p;   // LDS int4*

union Frag16 { uint4 q[2]; v16bf v; };

#if defined(__has_builtin)
#if __has_builtin(__builtin_amdgcn_cvt_pk_bf16_f32)
#define HAVE_PK_BF16 1
#endif
#if __has_builtin(__builtin_amdgcn_global_load_async_to_lds_b128)
#define HAVE_ASYNC_LDS 1
#endif
#if __has_builtin(__builtin_amdgcn_s_wait_asynccnt)
#define HAVE_WAIT_ASYNC 1
#endif
#endif

// Hardware packed f32->bf16 (RNE). Builtin if declared, else raw instruction.
__device__ __forceinline__ unsigned pack2(float a, float b) {
#ifdef HAVE_PK_BF16
  auto r = __builtin_amdgcn_cvt_pk_bf16_f32(a, b);
  return __builtin_bit_cast(unsigned, r);
#else
  unsigned r;
  asm("v_cvt_pk_bf16_f32 %0, %1, %2" : "=v"(r) : "v"(a), "v"(b));
  return r;
#endif
}

__device__ __forceinline__ void async_wait0() {
#ifdef HAVE_WAIT_ASYNC
  __builtin_amdgcn_s_wait_asynccnt(0);
#else
  asm volatile("s_wait_asynccnt 0x0" ::: "memory");
#endif
}

__device__ __forceinline__ v8f wmma_bf16(v16bf a, v16bf b, v8f c) {
  return __builtin_amdgcn_wmma_f32_16x16x32_bf16(false, a, false, b, (short)0, c,
                                                 false, false);
}

// A-matrix 16x32 bf16 fragment: lane half selects K offset 8 within 16-groups
__device__ __forceinline__ v16bf ldA(const unsigned short* p, int row, int ha) {
  Frag16 f;
  const unsigned short* r = p + row * 32;
  f.q[0] = *(const uint4*)(r + ha);
  f.q[1] = *(const uint4*)(r + 16 + ha);
  return f.v;
}
// B-matrix 32x16 bf16 fragment: lane = column, lane half selects K offset 8 within 16-groups
__device__ __forceinline__ v16bf ldB(const unsigned short* p, int col, int hb) {
  Frag16 f;
  const unsigned short* r = p + col * 32;
  f.q[0] = *(const uint4*)(r + hb);
  f.q[1] = *(const uint4*)(r + hb + 16);
  return f.v;
}

// ---------------------------------------------------------------- init
__global__ void k_init(int* cnt, int* cur) {
  int t = threadIdx.x;
  if (t < NEXP) { cnt[t] = 0; cur[t] = 0; }
}

// ---------------------------------------------------------------- router
__global__ __launch_bounds__(256) void k_router(
    const float* __restrict__ x, const float* __restrict__ Wg,
    const float* __restrict__ bg, int* __restrict__ idx,
    float* __restrict__ gate, float* __restrict__ part) {
  __shared__ float sWg[NEXP][D_MODEL];
  __shared__ float red[256];
  int t = threadIdx.x;
  for (int i = t; i < NEXP * D_MODEL; i += 256) sWg[i >> 10][i & 1023] = Wg[i];
  __syncthreads();

  int n = blockIdx.x * 256 + t;
  float acc[NEXP];
#pragma unroll
  for (int e = 0; e < NEXP; ++e) acc[e] = bg[e];
  const float4* xp = reinterpret_cast<const float4*>(x + (size_t)n * D_MODEL);
  for (int d4 = 0; d4 < D_MODEL / 4; ++d4) {
    float4 xv = xp[d4];
    int d = d4 * 4;
#pragma unroll
    for (int e = 0; e < NEXP; ++e)
      acc[e] += xv.x * sWg[e][d] + xv.y * sWg[e][d + 1] +
                xv.z * sWg[e][d + 2] + xv.w * sWg[e][d + 3];
  }
  // softmax + argmax (first max wins, like jnp.argmax)
  float mx = acc[0]; int bi = 0;
#pragma unroll
  for (int e = 1; e < NEXP; ++e) if (acc[e] > mx) { mx = acc[e]; bi = e; }
  float ps[NEXP]; float sum = 0.f;
#pragma unroll
  for (int e = 0; e < NEXP; ++e) { ps[e] = __expf(acc[e] - mx); sum += ps[e]; }
  float inv = 1.f / sum;
  idx[n] = bi;
  gate[n] = ps[bi] * inv;

  // deterministic per-block reductions: prob-sum and one-hot count per expert
  for (int e = 0; e < NEXP; ++e) {
    __syncthreads();
    red[t] = ps[e] * inv;
    __syncthreads();
    for (int s = 128; s > 0; s >>= 1) { if (t < s) red[t] += red[t + s]; __syncthreads(); }
    if (t == 0) part[blockIdx.x * 16 + e] = red[0];
    __syncthreads();
    red[t] = (bi == e) ? 1.f : 0.f;
    __syncthreads();
    for (int s = 128; s > 0; s >>= 1) { if (t < s) red[t] += red[t + s]; __syncthreads(); }
    if (t == 0) part[blockIdx.x * 16 + 8 + e] = red[0];
  }
}

// ---------------------------------------------------------------- aux loss
__global__ void k_aux(const float* __restrict__ part, float* __restrict__ out_aux) {
  __shared__ float s[16];
  int t = threadIdx.x;
  if (t < 16) {
    float a = 0.f;
    for (int b = 0; b < NTOK / 256; ++b) a += part[b * 16 + t];  // fixed order
    s[t] = a;
  }
  __syncthreads();
  if (t == 0) {
    float aux = 0.f;
    for (int e = 0; e < NEXP; ++e)
      aux += (s[8 + e] / (float)NTOK) * (s[e] / (float)NTOK);   // me * ce
    *out_aux = ALPHA_LB * (float)NEXP * aux;
  }
}

// ---------------------------------------------------------------- bucketize
__global__ void k_count(const int* __restrict__ idx, int* __restrict__ cnt) {
  int n = blockIdx.x * blockDim.x + threadIdx.x;
  if (n < NTOK) atomicAdd(&cnt[idx[n]], 1);
}
__global__ void k_scan(const int* __restrict__ cnt, int* __restrict__ offs,
                       int* __restrict__ cur) {
  if (threadIdx.x == 0) {
    int o = 0;
    for (int e = 0; e < NEXP; ++e) { offs[e] = o; cur[e] = o; o += cnt[e]; }
  }
}
__global__ void k_fill(const int* __restrict__ idx, int* __restrict__ cur,
                       int* __restrict__ perm) {
  int n = blockIdx.x * blockDim.x + threadIdx.x;
  if (n < NTOK) { int s = atomicAdd(&cur[idx[n]], 1); perm[s] = n; }
}

// ---------------------------------------------------------------- GEMM1: up + SwiGLU (+gate)
// Block tile: 128 tokens x 64 hidden; 8 waves as 4(m) x 2(n); wave tile 32x32
// for both U and V -> 8 WMMAs per k-step per wave.
__global__ __launch_bounds__(256) void k_up(
    const float* __restrict__ x, const float* __restrict__ Wu,
    const float* __restrict__ Wv, const int* __restrict__ perm,
    const int* __restrict__ cnt, const int* __restrict__ offs,
    const float* __restrict__ gate, unsigned short* __restrict__ hact) {
  int e = blockIdx.z;
  int mc = cnt[e];
  int mt = blockIdx.y;
  if (mt * 128 >= mc) return;
  int slot0   = offs[e] + mt * 128;
  int slotEnd = offs[e] + mc;
  int hBase   = blockIdx.x * 64;

  __shared__ unsigned short sA[128 * 32];
  __shared__ unsigned short sU[64 * 32];
  __shared__ unsigned short sV[64 * 32];

  int t = threadIdx.x, lane = t & 31, w = t >> 5;
  int m0 = (w & 3) * 32, n0 = (w >> 2) * 32;

  int arow = t >> 1, acol = (t & 1) * 16;       // A loader: 16 floats/thread
  int wrow = t >> 2, wcol = (t & 3) * 8;        // weight loaders: 8 floats/thread

  const float* WuB = Wu + (size_t)e * HID * D_MODEL + (size_t)hBase * D_MODEL;
  const float* WvB = Wv + (size_t)e * HID * D_MODEL + (size_t)hBase * D_MODEL;
  int tokA = (slot0 + arow < slotEnd) ? perm[slot0 + arow] : -1;

  v8f accU[2][2] = {};
  v8f accV[2][2] = {};
  int l15 = lane & 15;
  int ha = (lane >> 4) * 8;     // A-frag K sub-offset (elements)
  int hb = (lane >> 4) * 8;     // B-frag K sub-offset (elements)

  for (int k0 = 0; k0 < D_MODEL; k0 += 32) {
    __syncthreads();
    if (tokA >= 0) {
      const float4* p = reinterpret_cast<const float4*>(x + (size_t)tokA * D_MODEL + k0 + acol);
      float4 a = p[0], b = p[1], c = p[2], d = p[3];
      uint4* dst = (uint4*)&sA[arow * 32 + acol];
      dst[0] = make_uint4(pack2(a.x, a.y), pack2(a.z, a.w), pack2(b.x, b.y), pack2(b.z, b.w));
      dst[1] = make_uint4(pack2(c.x, c.y), pack2(c.z, c.w), pack2(d.x, d.y), pack2(d.z, d.w));
    } else {
      uint4* dst = (uint4*)&sA[arow * 32 + acol];
      dst[0] = make_uint4(0, 0, 0, 0);
      dst[1] = make_uint4(0, 0, 0, 0);
    }
    {
      const float* srcU = WuB + (size_t)wrow * D_MODEL + k0 + wcol;
      const float* srcV = WvB + (size_t)wrow * D_MODEL + k0 + wcol;
      const float4* p = reinterpret_cast<const float4*>(srcU);
      float4 a = p[0], b = p[1];
      *(uint4*)&sU[wrow * 32 + wcol] =
          make_uint4(pack2(a.x, a.y), pack2(a.z, a.w), pack2(b.x, b.y), pack2(b.z, b.w));
      const float4* q = reinterpret_cast<const float4*>(srcV);
      a = q[0]; b = q[1];
      *(uint4*)&sV[wrow * 32 + wcol] =
          make_uint4(pack2(a.x, a.y), pack2(a.z, a.w), pack2(b.x, b.y), pack2(b.z, b.w));
      if (k0 + 32 < D_MODEL) {                    // cover latency of next k-slice
        __builtin_prefetch(srcU + 32, 0, 0);
        __builtin_prefetch(srcV + 32, 0, 0);
      }
    }
    __syncthreads();

    v16bf fa0 = ldA(sA, m0 + l15, ha);
    v16bf fa1 = ldA(sA, m0 + 16 + l15, ha);
    v16bf fu0 = ldB(sU, n0 + l15, hb);
    v16bf fu1 = ldB(sU, n0 + 16 + l15, hb);
    v16bf fv0 = ldB(sV, n0 + l15, hb);
    v16bf fv1 = ldB(sV, n0 + 16 + l15, hb);

    accU[0][0] = wmma_bf16(fa0, fu0, accU[0][0]);
    accU[0][1] = wmma_bf16(fa0, fu1, accU[0][1]);
    accU[1][0] = wmma_bf16(fa1, fu0, accU[1][0]);
    accU[1][1] = wmma_bf16(fa1, fu1, accU[1][1]);
    accV[0][0] = wmma_bf16(fa0, fv0, accV[0][0]);
    accV[0][1] = wmma_bf16(fa0, fv1, accV[0][1]);
    accV[1][0] = wmma_bf16(fa1, fv0, accV[1][0]);
    accV[1][1] = wmma_bf16(fa1, fv1, accV[1][1]);
  }

  // epilogue: silu(u) * v * gate -> bf16 h_act (grouped by slot)
  int hcol = hBase + n0;
#pragma unroll
  for (int s = 0; s < 2; ++s) {
    int rbase = m0 + s * 16 + ((lane >> 4) << 3);
#pragma unroll
    for (int r = 0; r < 8; ++r) {
      int slot = slot0 + rbase + r;
      if (slot < slotEnd) {
        int tok = perm[slot];
        float g = gate[tok];
        float u0 = accU[s][0][r], v0 = accV[s][0][r];
        float u1 = accU[s][1][r], v1 = accV[s][1][r];
        float s0 = (u0 / (1.f + __expf(-u0))) * v0 * g;
        float s1 = (u1 / (1.f + __expf(-u1))) * v1 * g;
        unsigned pk = pack2(s0, s1);
        size_t ob = (size_t)slot * HID + hcol;
        hact[ob + l15]      = (unsigned short)pk;
        hact[ob + 16 + l15] = (unsigned short)(pk >> 16);
      }
    }
  }
}

// ---------------------------------------------------------------- GEMM2: down + scatter
// Block tile: 128 slots x 64 d_model; wave tile 32x32 -> 4 WMMAs per k-step.
// A-tile (already bf16) is copied global->LDS with the gfx1250 async path.
__global__ __launch_bounds__(256) void k_down(
    const unsigned short* __restrict__ hact, const float* __restrict__ Wd,
    const int* __restrict__ perm, const int* __restrict__ cnt,
    const int* __restrict__ offs, float* __restrict__ y) {
  int e = blockIdx.z;
  int mc = cnt[e];
  int mt = blockIdx.y;
  if (mt * 128 >= mc) return;
  int slot0   = offs[e] + mt * 128;
  int slotEnd = offs[e] + mc;
  int dBase   = blockIdx.x * 64;

  __shared__ unsigned short sA[128 * 32];
  __shared__ unsigned short sB[64 * 32];

  int t = threadIdx.x, lane = t & 31, w = t >> 5;
  int m0 = (w & 3) * 32, n0 = (w >> 2) * 32;

  int arow = t >> 1, acol = (t & 1) * 16;
  int wrow = t >> 2, wcol = (t & 3) * 8;

  const float* WdB = Wd + (size_t)e * D_MODEL * HID + (size_t)dBase * HID;
  bool rowValid = (slot0 + arow < slotEnd);
  const unsigned short* aSrc = hact + (size_t)(slot0 + arow) * HID + acol;

  v8f acc[2][2] = {};
  int l15 = lane & 15;
  int ha = (lane >> 4) * 8;
  int hb = (lane >> 4) * 8;

  for (int k0 = 0; k0 < HID; k0 += 32) {
    __syncthreads();
    {
      uint4* dst = (uint4*)&sA[arow * 32 + acol];
      if (rowValid) {
#ifdef HAVE_ASYNC_LDS
        __builtin_amdgcn_global_load_async_to_lds_b128(
            (gv4i_p)(aSrc + k0), (lv4i_p)(dst), 0, 0);
        __builtin_amdgcn_global_load_async_to_lds_b128(
            (gv4i_p)(aSrc + k0 + 8), (lv4i_p)(dst + 1), 0, 0);
#else
        const uint4* p = reinterpret_cast<const uint4*>(aSrc + k0);
        dst[0] = p[0];
        dst[1] = p[1];
#endif
      } else {
        dst[0] = make_uint4(0, 0, 0, 0);
        dst[1] = make_uint4(0, 0, 0, 0);
      }
    }
    {
      const float* srcB = WdB + (size_t)wrow * HID + k0 + wcol;
      const float4* p = reinterpret_cast<const float4*>(srcB);
      float4 a = p[0], b = p[1];
      *(uint4*)&sB[wrow * 32 + wcol] =
          make_uint4(pack2(a.x, a.y), pack2(a.z, a.w), pack2(b.x, b.y), pack2(b.z, b.w));
      if (k0 + 32 < HID) __builtin_prefetch(srcB + 32, 0, 0);
    }
#ifdef HAVE_ASYNC_LDS
    async_wait0();                 // async tile complete before barrier
#endif
    __syncthreads();

    v16bf fa0 = ldA(sA, m0 + l15, ha);
    v16bf fa1 = ldA(sA, m0 + 16 + l15, ha);
    v16bf fb0 = ldB(sB, n0 + l15, hb);
    v16bf fb1 = ldB(sB, n0 + 16 + l15, hb);

    acc[0][0] = wmma_bf16(fa0, fb0, acc[0][0]);
    acc[0][1] = wmma_bf16(fa0, fb1, acc[0][1]);
    acc[1][0] = wmma_bf16(fa1, fb0, acc[1][0]);
    acc[1][1] = wmma_bf16(fa1, fb1, acc[1][1]);
  }

  int dcol = dBase + n0;
#pragma unroll
  for (int s = 0; s < 2; ++s) {
    int rbase = m0 + s * 16 + ((lane >> 4) << 3);
#pragma unroll
    for (int r = 0; r < 8; ++r) {
      int slot = slot0 + rbase + r;
      if (slot < slotEnd) {
        int tok = perm[slot];
        size_t ob = (size_t)tok * D_MODEL + dcol;
        y[ob + l15]      = acc[s][0][r];
        y[ob + 16 + l15] = acc[s][1][r];
      }
    }
  }
}

// ---------------------------------------------------------------- launch
extern "C" void kernel_launch(void* const* d_in, const int* in_sizes, int n_in,
                              void* d_out, int out_size, void* d_ws, size_t ws_size,
                              hipStream_t stream) {
  (void)in_sizes; (void)n_in; (void)out_size; (void)ws_size;
  const float* x  = (const float*)d_in[0];
  const float* Wg = (const float*)d_in[1];
  const float* bg = (const float*)d_in[2];
  const float* Wu = (const float*)d_in[3];
  const float* Wv = (const float*)d_in[4];
  const float* Wd = (const float*)d_in[5];
  float* y = (float*)d_out;

  // workspace layout (bytes)
  char* ws = (char*)d_ws;
  float* gate = (float*)(ws);                       // N floats
  int*   idx  = (int*)(ws + (size_t)NTOK * 4);      // N ints
  int*   cnt  = (int*)(ws + (size_t)2 * NTOK * 4);  // 8
  int*   cur  = (int*)(ws + (size_t)2 * NTOK * 4 + 32);
  int*   offs = (int*)(ws + (size_t)2 * NTOK * 4 + 64);
  float* part = (float*)(ws + (size_t)2 * NTOK * 4 + 96);            // 32*16 floats
  int*   perm = (int*)(ws + (size_t)2 * NTOK * 4 + 96 + 2048);       // N ints
  size_t hoff = ((size_t)2 * NTOK * 4 + 96 + 2048 + (size_t)NTOK * 4 + 255) & ~(size_t)255;
  unsigned short* hact = (unsigned short*)(ws + hoff);               // N*H bf16 (~45 MB)

  k_init<<<1, 32, 0, stream>>>(cnt, cur);
  k_router<<<NTOK / 256, 256, 0, stream>>>(x, Wg, bg, idx, gate, part);
  k_aux<<<1, 64, 0, stream>>>(part, y + (size_t)NTOK * D_MODEL);
  k_count<<<NTOK / 256, 256, 0, stream>>>(idx, cnt);
  k_scan<<<1, 32, 0, stream>>>(cnt, offs, cur);
  k_fill<<<NTOK / 256, 256, 0, stream>>>(idx, cur, perm);

  dim3 g1(HID / 64, NTOK / 128, NEXP);
  k_up<<<g1, 256, 0, stream>>>(x, Wu, Wv, perm, cnt, offs, gate, hact);
  dim3 g2(D_MODEL / 64, NTOK / 128, NEXP);
  k_down<<<g2, 256, 0, stream>>>(hact, Wd, perm, cnt, offs, y);
}
